// MultiHeadSelfAttentionWithLagBias_57389353009846
// MI455X (gfx1250) — matrix operand order repaired and checked by reading, not compiled, not run on hardware
//
#include <hip/hip_runtime.h>
#include <hip/hip_bf16.h>
#include <stdint.h>

typedef __bf16 bf16_t;
typedef __attribute__((ext_vector_type(16))) __bf16    bf16x16;
typedef __attribute__((ext_vector_type(8)))  __bf16    bf16x8;
typedef __attribute__((ext_vector_type(8)))  float     floatx8;
typedef __attribute__((ext_vector_type(4)))  uint32_t  u32x4;
typedef __attribute__((ext_vector_type(8)))  uint32_t  u32x8;

#define D_MODEL 1024
#define N_HEADS 16
#define D_K     64
#define SEQ     2048
#define BATCH   2
#define MAX_LAG 512
#define TOKENS  (BATCH * SEQ)   // 4096

// ---------------------------------------------------------------------------
// LDS operand fetch helpers matching CDNA5 WMMA bf16 VGPR layouts (wave32).
// A (16x32, MxK): lane L -> row L&15; lanes<16 hold K {0..7,16..23},
//                 lanes>=16 hold K {8..15,24..31}. Two 16B loads.
// B (32x16, KxN): lane L -> col L&15; lanes<16 hold K {0..7,16..23},
//                 lanes>=16 hold K {8..15,24..31}. Two 16B loads from BT.
// ---------------------------------------------------------------------------
__device__ __forceinline__ bf16x16 lds_load_a(const bf16_t* base, int row, int ld,
                                              int k0, int lane) {
  const int half = (lane >> 4) & 1;
  const bf16_t* p = base + row * ld + k0 + half * 8;
  bf16x8 lo = *(const bf16x8*)(p);
  bf16x8 hi = *(const bf16x8*)(p + 16);
  bf16x16 a;
#pragma unroll
  for (int i = 0; i < 8; ++i) { a[i] = lo[i]; a[i + 8] = hi[i]; }
  return a;
}

__device__ __forceinline__ bf16x16 lds_load_b(const bf16_t* base, int col, int ld,
                                              int k0, int lane) {
  const int half = (lane >> 4) & 1;
  const bf16_t* p = base + col * ld + k0 + half * 8;
  bf16x8 lo = *(const bf16x8*)(p);
  bf16x8 hi = *(const bf16x8*)(p + 16);
  bf16x16 b;
#pragma unroll
  for (int i = 0; i < 8; ++i) { b[i] = lo[i]; b[i + 8] = hi[i]; }
  return b;
}

// ---------------------------------------------------------------------------
// Tensor Data Mover: 2D bf16 tile load Global -> LDS, with LDS row padding
// of 4 DWORDs (8 bf16) after every 16 DWORDs (32 bf16) => LDS row pitch 40.
// Descriptor bit layout per CDNA5 ISA sect. 8.3/8.4.
// Issued by one wave; tracked with TENSORcnt.
// ---------------------------------------------------------------------------
__device__ __forceinline__ void tdm_load_2d_bf16(uint32_t lds_byte_addr,
                                                 const void* gptr,
                                                 uint32_t tensor_d0,
                                                 uint32_t tensor_d1,
                                                 uint32_t d0_stride,
                                                 uint32_t tile_d0,
                                                 uint32_t tile_d1) {
  uint64_t ga = (uint64_t)(uintptr_t)gptr;
  u32x4 g0;
  g0.x = 1u;                                    // count=1 valid descriptor
  g0.y = lds_byte_addr;                         // lds_addr[63:32]
  g0.z = (uint32_t)ga;                          // global_addr[95:64]
  g0.w = (uint32_t)(ga >> 32) | (2u << 30);     // global_addr hi | type=2
  u32x8 g1;
  g1.s0 = (1u << 16)        // data_size = 1 (2 bytes)
        | (1u << 20)        // pad_enable
        | (3u << 22)        // pad_interval: 16 DWORDs
        | (3u << 25);       // pad_amount: 4 DWORDs
  g1.s1 = (tensor_d0 & 0xffffu) << 16;                       // tensor_dim0 lo
  g1.s2 = (tensor_d0 >> 16) | ((tensor_d1 & 0xffffu) << 16); // dim0 hi|dim1 lo
  g1.s3 = (tensor_d1 >> 16) | (tile_d0 << 16);               // dim1 hi|tile0
  g1.s4 = tile_d1;                                           // tile1, tile2=0
  g1.s5 = d0_stride;                                         // dim0 stride lo
  g1.s6 = 0u;                                                // stride hi
  g1.s7 = 0u;
  asm volatile("tensor_load_to_lds %0, %1" :: "s"(g0), "s"(g1) : "memory");
}

// ---------------------------------------------------------------------------
// fp32 -> bf16 convert
// ---------------------------------------------------------------------------
__global__ void cvt_f32_bf16(const float* __restrict__ in, bf16_t* __restrict__ out,
                             int n) {
  int i = blockIdx.x * blockDim.x + threadIdx.x;
  if (i < n) out[i] = (bf16_t)in[i];
}

// fp32 W[k][n] -> bf16 WT[n][k]  (1024x1024)
__global__ void transpose_w_bf16(const float* __restrict__ w, bf16_t* __restrict__ wt) {
  int idx = blockIdx.x * blockDim.x + threadIdx.x;   // 0 .. 1M-1
  int k = idx >> 10;
  int n = idx & 1023;
  wt[n * D_MODEL + k] = (bf16_t)w[idx];
}

// ---------------------------------------------------------------------------
// C[4096x1024] = A[4096x1024](bf16) @ BT[1024(n) x 1024(k)](bf16)^T + bias
// mode 0: store bf16 per-head scatter [B,H,S,dk]
// mode 1: store fp32 row-major (final output)
// Block tile 64x128, 8 waves, each wave 16x64 (4 accumulators).
// Tile staging via TDM (double-buffered), waves only ds_load + wmma.
// ---------------------------------------------------------------------------
__global__ __launch_bounds__(256)
void gemm_bf16_wmma(const bf16_t* __restrict__ A, const bf16_t* __restrict__ BT,
                    const float* __restrict__ bias, void* __restrict__ Cout,
                    int mode) {
  __shared__ __align__(16) bf16_t As[2][64 * 40];    // 64 x 32, pitch 40
  __shared__ __align__(16) bf16_t Bs[2][128 * 40];   // 128 x 32, pitch 40

  const int tid   = threadIdx.x;
  const int lane  = tid & 31;
  const int wid   = tid >> 5;
  const int waveM = wid & 3;        // 4 M stripes
  const int waveN = wid >> 2;       // 2 N halves
  const int m0 = blockIdx.y * 64;
  const int n0 = blockIdx.x * 128;
  const int K  = D_MODEL;

  const bf16_t* gA = A + (size_t)m0 * K;    // tile row origin, k added below
  const bf16_t* gB = BT + (size_t)n0 * K;

  const uint32_t ldsA0 = (uint32_t)(uintptr_t)&As[0][0];
  const uint32_t ldsA1 = (uint32_t)(uintptr_t)&As[1][0];
  const uint32_t ldsB0 = (uint32_t)(uintptr_t)&Bs[0][0];
  const uint32_t ldsB1 = (uint32_t)(uintptr_t)&Bs[1][0];

  floatx8 acc[4];
#pragma unroll
  for (int t = 0; t < 4; ++t)
#pragma unroll
    for (int i = 0; i < 8; ++i) acc[t][i] = 0.0f;

  // prologue: fill buffer 0
  if (wid == 0) {
    tdm_load_2d_bf16(ldsA0, gA, K, 4096, K, 32, 64);
    tdm_load_2d_bf16(ldsB0, gB, K, 4096, K, 32, 128);
  }

  for (int k0 = 0; k0 < K; k0 += 32) {
    const int cur = (k0 >> 5) & 1;
    if (wid == 0) __builtin_amdgcn_s_wait_tensorcnt(0);  // cur buffer landed
    __syncthreads();
    if (wid == 0 && (k0 + 32) < K) {                     // prefetch next buffer
      tdm_load_2d_bf16(cur ? ldsA0 : ldsA1, gA + k0 + 32, K, 4096, K, 32, 64);
      tdm_load_2d_bf16(cur ? ldsB0 : ldsB1, gB + k0 + 32, K, 4096, K, 32, 128);
    }

    // fetch all operands into distinct registers first so the compiler can
    // use partial DS-count waits and overlap ds_loads with WMMA issue
    bf16x16 a = lds_load_a(&As[cur][0], waveM * 16 + (lane & 15), 40, 0, lane);
    bf16x16 b0 = lds_load_b(&Bs[cur][0], waveN * 64 +  0 + (lane & 15), 40, 0, lane);
    bf16x16 b1 = lds_load_b(&Bs[cur][0], waveN * 64 + 16 + (lane & 15), 40, 0, lane);
    bf16x16 b2 = lds_load_b(&Bs[cur][0], waveN * 64 + 32 + (lane & 15), 40, 0, lane);
    bf16x16 b3 = lds_load_b(&Bs[cur][0], waveN * 64 + 48 + (lane & 15), 40, 0, lane);
    acc[0] = __builtin_amdgcn_wmma_f32_16x16x32_bf16(false, a, false, b0, (short)0, acc[0], false, false);
    acc[1] = __builtin_amdgcn_wmma_f32_16x16x32_bf16(false, a, false, b1, (short)0, acc[1], false, false);
    acc[2] = __builtin_amdgcn_wmma_f32_16x16x32_bf16(false, a, false, b2, (short)0, acc[2], false, false);
    acc[3] = __builtin_amdgcn_wmma_f32_16x16x32_bf16(false, a, false, b3, (short)0, acc[3], false, false);
  }

  // epilogue: C layout -> VGPR r holds M=r (lanes 0-15) / M=r+8 (lanes 16-31)
  const int nloc  = lane & 15;
  const int mhalf = (lane >> 4) * 8;
#pragma unroll
  for (int t = 0; t < 4; ++t) {
    int n = n0 + waveN * 64 + t * 16 + nloc;
    float bn = bias[n];
#pragma unroll
    for (int r = 0; r < 8; ++r) {
      int m = m0 + waveM * 16 + mhalf + r;
      float v = acc[t][r] + bn;
      if (mode == 0) {
        int bb = m >> 11, s = m & 2047, h = n >> 6, d = n & 63;
        ((bf16_t*)Cout)[(((bb * N_HEADS + h) * SEQ) + s) * D_K + d] = (bf16_t)v;
      } else {
        ((float*)Cout)[m * D_MODEL + n] = v;
      }
    }
  }
}

// ---------------------------------------------------------------------------
// Flash attention with lag bias. One block = (b, h, 64-query tile).
// 4 waves, each wave owns a 16-query stripe. dk = 64 -> 2 WMMA K-steps.
// ---------------------------------------------------------------------------
__global__ __launch_bounds__(128)
void attn_kernel(const bf16_t* __restrict__ Q, const bf16_t* __restrict__ Km,
                 const bf16_t* __restrict__ V, const int* __restrict__ lag,
                 const float* __restrict__ lag_bias, bf16_t* __restrict__ out) {
  __shared__ __align__(16) bf16_t q_s[64 * 72];   // queries row-major (A src)
  __shared__ __align__(16) bf16_t k_s[64 * 72];   // keys row-major == BT(QK^T)
  __shared__ __align__(16) bf16_t vt_s[64 * 72];  // V^T [d][key] == BT(P.V)
  __shared__ __align__(16) bf16_t p_s[64 * 72];   // probabilities (A src)
  __shared__ float  bias_s[MAX_LAG + 1];
  __shared__ int    lagq_s[64];
  __shared__ int    lagk_s[64];

  const int tid  = threadIdx.x;
  const int lane = tid & 31;
  const int w    = tid >> 5;          // wave 0..3
  const int qb   = blockIdx.x;        // 0..31
  const int h    = blockIdx.y;        // 0..15
  const int b    = blockIdx.z;        // 0..1
  const int q0   = qb * 64;

  const bf16_t* qbase = Q  + (size_t)((b * N_HEADS + h) * SEQ) * D_K;
  const bf16_t* kbase = Km + (size_t)((b * N_HEADS + h) * SEQ) * D_K;
  const bf16_t* vbase = V  + (size_t)((b * N_HEADS + h) * SEQ) * D_K;

  // stage Q, lagq, bias row
#pragma unroll
  for (int r = 0; r < 4; ++r) {
    int e   = tid + r * 128;          // 64 rows x 8 groups of 8
    int row = e >> 3;
    int cg  = (e & 7) * 8;
    *(bf16x8*)&q_s[row * 72 + cg] = *(const bf16x8*)&qbase[(q0 + row) * D_K + cg];
  }
  if (tid < 64) lagq_s[tid] = lag[q0 + tid];
  for (int i = tid; i < MAX_LAG + 1; i += 128)
    bias_s[i] = lag_bias[h * (MAX_LAG + 1) + i];

  float   m_i[8], l_i[8];
  floatx8 acc_o[4];
#pragma unroll
  for (int r = 0; r < 8; ++r) { m_i[r] = -3.0e38f; l_i[r] = 0.0f; }
#pragma unroll
  for (int t = 0; t < 4; ++t)
#pragma unroll
    for (int i = 0; i < 8; ++i) acc_o[t][i] = 0.0f;

  const int nloc    = lane & 15;
  const int rowbase = w * 16 + (lane >> 4) * 8;  // first of this lane's 8 rows

  for (int kb = 0; kb < SEQ; kb += 64) {
    __syncthreads();   // previous iteration done reading k_s / vt_s
    // stage K row-major and V transposed, plus key lags
#pragma unroll
    for (int r = 0; r < 4; ++r) {
      int e   = tid + r * 128;
      int row = e >> 3;
      int cg  = (e & 7) * 8;
      *(bf16x8*)&k_s[row * 72 + cg] = *(const bf16x8*)&kbase[(kb + row) * D_K + cg];
      bf16x8 vv = *(const bf16x8*)&vbase[(kb + row) * D_K + cg];
#pragma unroll
      for (int j = 0; j < 8; ++j) vt_s[(cg + j) * 72 + row] = vv[j];
    }
    if (tid < 64) lagk_s[tid] = lag[kb + tid];
    // prefetch next K/V tiles into cache while we compute this one
    if (kb + 64 < SEQ) {
      __builtin_prefetch(&kbase[(size_t)(kb + 64) * D_K], 0, 3);
      __builtin_prefetch(&vbase[(size_t)(kb + 64) * D_K], 0, 3);
    }
    __syncthreads();

    // S = Q @ K^T   (16 queries x 64 keys per wave)
    floatx8 sacc[4];
#pragma unroll
    for (int t = 0; t < 4; ++t)
#pragma unroll
      for (int i = 0; i < 8; ++i) sacc[t][i] = 0.0f;
#pragma unroll
    for (int kk = 0; kk < 2; ++kk) {
      bf16x16 a  = lds_load_a(q_s, w * 16 + nloc, 72, kk * 32, lane);
      bf16x16 b0 = lds_load_b(k_s,  0 + nloc, 72, kk * 32, lane);
      bf16x16 b1 = lds_load_b(k_s, 16 + nloc, 72, kk * 32, lane);
      bf16x16 b2 = lds_load_b(k_s, 32 + nloc, 72, kk * 32, lane);
      bf16x16 b3 = lds_load_b(k_s, 48 + nloc, 72, kk * 32, lane);
      sacc[0] = __builtin_amdgcn_wmma_f32_16x16x32_bf16(false, a, false, b0, (short)0, sacc[0], false, false);
      sacc[1] = __builtin_amdgcn_wmma_f32_16x16x32_bf16(false, a, false, b1, (short)0, sacc[1], false, false);
      sacc[2] = __builtin_amdgcn_wmma_f32_16x16x32_bf16(false, a, false, b2, (short)0, sacc[2], false, false);
      sacc[3] = __builtin_amdgcn_wmma_f32_16x16x32_bf16(false, a, false, b3, (short)0, sacc[3], false, false);
    }

    // scale + lag bias, row max
    float sv[4][8];
    float rmax[8];
#pragma unroll
    for (int r = 0; r < 8; ++r) rmax[r] = -3.0e38f;
#pragma unroll
    for (int t = 0; t < 4; ++t) {
      int lk = lagk_s[t * 16 + nloc];
#pragma unroll
      for (int r = 0; r < 8; ++r) {
        int lq = lagq_s[rowbase + r];
        int ld = lq - lk; if (ld < 0) ld = -ld;
        float s = sacc[t][r] * 0.125f + bias_s[ld];
        sv[t][r] = s;
        rmax[r] = fmaxf(rmax[r], s);
      }
    }
    // butterfly max over the 16 lanes holding one row
#pragma unroll
    for (int r = 0; r < 8; ++r) {
      rmax[r] = fmaxf(rmax[r], __shfl_xor(rmax[r], 1, 32));
      rmax[r] = fmaxf(rmax[r], __shfl_xor(rmax[r], 2, 32));
      rmax[r] = fmaxf(rmax[r], __shfl_xor(rmax[r], 4, 32));
      rmax[r] = fmaxf(rmax[r], __shfl_xor(rmax[r], 8, 32));
    }

    // online softmax update
#pragma unroll
    for (int r = 0; r < 8; ++r) {
      float mnew = fmaxf(m_i[r], rmax[r]);
      float corr = __expf(m_i[r] - mnew);
      float psum = 0.0f;
#pragma unroll
      for (int t = 0; t < 4; ++t) {
        float p = __expf(sv[t][r] - mnew);
        sv[t][r] = p;
        psum += p;
      }
      psum += __shfl_xor(psum, 1, 32);
      psum += __shfl_xor(psum, 2, 32);
      psum += __shfl_xor(psum, 4, 32);
      psum += __shfl_xor(psum, 8, 32);
      l_i[r] = l_i[r] * corr + psum;
      m_i[r] = mnew;
#pragma unroll
      for (int t = 0; t < 4; ++t) acc_o[t][r] *= corr;
    }

    // P -> LDS (re-layout C-shape -> row-major A-operand source)
#pragma unroll
    for (int t = 0; t < 4; ++t)
#pragma unroll
      for (int r = 0; r < 8; ++r)
        p_s[(rowbase + r) * 72 + t * 16 + nloc] = (bf16_t)sv[t][r];

    __syncthreads();

    // O += P @ V
#pragma unroll
    for (int kk = 0; kk < 2; ++kk) {
      bf16x16 a  = lds_load_a(p_s, w * 16 + nloc, 72, kk * 32, lane);
      bf16x16 b0 = lds_load_b(vt_s,  0 + nloc, 72, kk * 32, lane);
      bf16x16 b1 = lds_load_b(vt_s, 16 + nloc, 72, kk * 32, lane);
      bf16x16 b2 = lds_load_b(vt_s, 32 + nloc, 72, kk * 32, lane);
      bf16x16 b3 = lds_load_b(vt_s, 48 + nloc, 72, kk * 32, lane);
      acc_o[0] = __builtin_amdgcn_wmma_f32_16x16x32_bf16(false, a, false, b0, (short)0, acc_o[0], false, false);
      acc_o[1] = __builtin_amdgcn_wmma_f32_16x16x32_bf16(false, a, false, b1, (short)0, acc_o[1], false, false);
      acc_o[2] = __builtin_amdgcn_wmma_f32_16x16x32_bf16(false, a, false, b2, (short)0, acc_o[2], false, false);
      acc_o[3] = __builtin_amdgcn_wmma_f32_16x16x32_bf16(false, a, false, b3, (short)0, acc_o[3], false, false);
    }
  }

  // normalize and scatter to [tokens x D_MODEL] bf16
#pragma unroll
  for (int r = 0; r < 8; ++r) {
    float inv = 1.0f / l_i[r];
    int tok = b * SEQ + q0 + rowbase + r;
#pragma unroll
    for (int t = 0; t < 4; ++t) {
      int d = t * 16 + nloc;
      out[(size_t)tok * D_MODEL + h * D_K + d] = (bf16_t)(acc_o[t][r] * inv);
    }
  }
}

// ---------------------------------------------------------------------------
extern "C" void kernel_launch(void* const* d_in, const int* in_sizes, int n_in,
                              void* d_out, int out_size, void* d_ws, size_t ws_size,
                              hipStream_t stream) {
  (void)in_sizes; (void)n_in; (void)out_size; (void)ws_size;
  const float* x        = (const float*)d_in[0];
  const int*   lag      = (const int*)d_in[1];
  const float* wq       = (const float*)d_in[2];
  const float* bq       = (const float*)d_in[3];
  const float* wk       = (const float*)d_in[4];
  const float* bk       = (const float*)d_in[5];
  const float* wv       = (const float*)d_in[6];
  const float* bv       = (const float*)d_in[7];
  const float* wo       = (const float*)d_in[8];
  const float* bo       = (const float*)d_in[9];
  const float* lag_bias = (const float*)d_in[10];
  float* out = (float*)d_out;

  char* ws = (char*)d_ws;
  const size_t MB = 1024u * 1024u;
  bf16_t* xbf = (bf16_t*)(ws + 0 * MB);   // 8 MiB  [4096 x 1024]
  bf16_t* wqT = (bf16_t*)(ws + 8 * MB);   // 2 MiB  [n][k]
  bf16_t* wkT = (bf16_t*)(ws + 10 * MB);
  bf16_t* wvT = (bf16_t*)(ws + 12 * MB);
  bf16_t* woT = (bf16_t*)(ws + 14 * MB);
  bf16_t* Qh  = (bf16_t*)(ws + 16 * MB);  // 8 MiB  [B,H,S,dk]
  bf16_t* Kh  = (bf16_t*)(ws + 24 * MB);
  bf16_t* Vh  = (bf16_t*)(ws + 32 * MB);
  bf16_t* Ao  = (bf16_t*)(ws + 40 * MB);  // 8 MiB  [4096 x 1024]

  cvt_f32_bf16<<<(TOKENS * D_MODEL) / 256, 256, 0, stream>>>(x, xbf, TOKENS * D_MODEL);
  transpose_w_bf16<<<(D_MODEL * D_MODEL) / 256, 256, 0, stream>>>(wq, wqT);
  transpose_w_bf16<<<(D_MODEL * D_MODEL) / 256, 256, 0, stream>>>(wk, wkT);
  transpose_w_bf16<<<(D_MODEL * D_MODEL) / 256, 256, 0, stream>>>(wv, wvT);
  transpose_w_bf16<<<(D_MODEL * D_MODEL) / 256, 256, 0, stream>>>(wo, woT);

  dim3 gg(D_MODEL / 128, TOKENS / 64);    // (8, 64)
  gemm_bf16_wmma<<<gg, 256, 0, stream>>>(xbf, wqT, bq, (void*)Qh, 0);
  gemm_bf16_wmma<<<gg, 256, 0, stream>>>(xbf, wkT, bk, (void*)Kh, 0);
  gemm_bf16_wmma<<<gg, 256, 0, stream>>>(xbf, wvT, bv, (void*)Vh, 0);

  attn_kernel<<<dim3(SEQ / 64, N_HEADS, BATCH), 128, 0, stream>>>(
      Qh, Kh, Vh, lag, lag_bias, Ao);

  gemm_bf16_wmma<<<gg, 256, 0, stream>>>(Ao, woT, bo, (void*)out, 1);
}
